// HeteroFloodGNN_86062554677665
// MI455X (gfx1250) — matrix operand order repaired and checked by reading, not compiled, not run on hardware
//
#include <hip/hip_runtime.h>

// ---------------------------------------------------------------------------
// HeteroFloodGNN forward for MI455X (gfx1250, wave32, WMMA).
// Dense 64-wide projections use v_wmma_f32_16x16x32_bf16 (f32 accumulate),
// weights staged in LDS pre-swizzled into the WMMA B-fragment layout so each
// wave fetches its fragment as one contiguous 32B ds_load. A-fragments load
// as unconditional 128-bit global loads. The two K-steps use independent
// accumulators (no WMMA->WMMA RAW hazard NOPs), summed in the epilogue.
// All row counts (8000 / 200000) are exact multiples of the block tiles,
// so epilogue stores are unconditional. Scatter-softmax attention via 3
// atomic passes (uint-encoded atomicMax, f32 atomicAdd).
// ---------------------------------------------------------------------------

typedef __bf16  v16bf __attribute__((ext_vector_type(16)));
typedef float   v8f   __attribute__((ext_vector_type(8)));
typedef float   v4f   __attribute__((ext_vector_type(4)));

#define HID   64
#define HEADS 16
#define HC    4
#define EMB_D 16
#define KOUT  4
#define N1C   2000
#define N2C   50000
#define BATCH 4
#define N1T   (BATCH * N1C)    /* 8000   */
#define N2T   (BATCH * N2C)    /* 200000 */
#define E_PIPE_C 16000
#define E_SURF_C 800000
#define E_CC     8000

__device__ __forceinline__ float gelu_f(float x) {
    return 0.5f * x * (1.0f + erff(x * 0.70710678118654752440f));
}

// monotonic float <-> uint key for atomicMax-based segment max
__device__ __forceinline__ unsigned fkey(float f) {
    unsigned u = __float_as_uint(f);
    return (u >> 31) ? ~u : (u | 0x80000000u);
}
__device__ __forceinline__ float funkey(unsigned k) {
    return __uint_as_float((k >> 31) ? (k ^ 0x80000000u) : ~k);
}

// ---------------------------------------------------------------------------
// WMMA linear: Y[N x DOUT] = X[N x DIN] @ W[DIN x DOUT] + b
// One wave -> 16x16 tile; block = 8 waves (256 thr). Requires
// N % (16 * (8 / (DOUT/16))) == 0 (grid covers N exactly).
// A layout (16-bit A 16x32, contiguous K like B): lane&15 = row M, lane>>4 selects K
// 0..15 vs 16..31. B: lane&15 = col N, lane>>4 selects
// K 0..15 vs 16..31, VGPR j holds K=2j,2j+1. D: vgpr j -> M = j + 8*(lane>>4),
// N = lane&15. W is staged in LDS pre-swizzled per-fragment:
//   sw[((kb*CT + ct)*32 + lane)*16 + i] = bf16(W[(kb*32 + (lane>>4)*16 + i)*DOUT
//                                                + ct*16 + (lane&15)])
// so the inner loop does a single contiguous 32-byte LDS read per K-step.
// Two independent accumulators avoid the 5-cycle bf16 WMMA->WMMA RAW stall.
// ---------------------------------------------------------------------------
template <int DIN, int DOUT>
__global__ void linear_wmma_kernel(const float* __restrict__ X,
                                   const float* __restrict__ W,
                                   const float* __restrict__ bias,
                                   float* __restrict__ Y) {
    constexpr int CT = DOUT / 16;   // col tiles per block
    constexpr int RT = 8 / CT;      // row tiles per block
    constexpr int KB = DIN / 32;    // K steps
    static_assert(KB >= 1 && KB <= 4, "K depth");
    __shared__ alignas(32) __bf16 sw[DIN * DOUT];
    __shared__ float sb[DOUT];
    const int tid = threadIdx.x;
    for (int idx = tid; idx < DIN * DOUT; idx += 256) {
        const int i  = idx & 15;
        const int ln = (idx >> 4) & 31;
        const int ct = (idx >> 9) % CT;
        const int kb = idx / (512 * CT);
        const int k  = kb * 32 + (ln >> 4) * 16 + i;
        const int c  = ct * 16 + (ln & 15);
        sw[idx] = (__bf16)W[k * DOUT + c];
    }
    if (tid < DOUT) sb[tid] = bias[tid];
    __syncthreads();

    const int w    = tid >> 5;
    const int lane = tid & 31;
    const int r    = lane & 15;
    const int hi   = lane >> 4;
    const int R0   = (blockIdx.x * RT + (w / CT)) * 16;
    const int C0   = (w % CT) * 16;
    const int col  = C0 + r;

    const float* xr = X + (size_t)(R0 + r) * DIN;
    __builtin_prefetch(xr + (size_t)32 * DIN, 0, 1);   // stream-ahead hint

    const float bc = sb[col];
    v8f acc[2];
    acc[0] = (v8f){bc, bc, bc, bc, bc, bc, bc, bc};    // bias folded into C of step 0
    acc[1] = (v8f){0.f, 0.f, 0.f, 0.f, 0.f, 0.f, 0.f, 0.f};

#pragma unroll
    for (int kb = 0; kb < KB; ++kb) {
        const int k0 = kb * 32;
        const v4f a0 = *(const v4f*)(xr + k0 + hi * 16);
        const v4f a1 = *(const v4f*)(xr + k0 + hi * 16 + 4);
        const v4f a2 = *(const v4f*)(xr + k0 + hi * 16 + 8);
        const v4f a3 = *(const v4f*)(xr + k0 + hi * 16 + 12);
        v16bf av;
#pragma unroll
        for (int i = 0; i < 4; ++i) {
            av[i]      = (__bf16)a0[i];
            av[4 + i]  = (__bf16)a1[i];
            av[8 + i]  = (__bf16)a2[i];
            av[12 + i] = (__bf16)a3[i];
        }
        const v16bf bv = *(const v16bf*)(sw + ((kb * CT + (C0 >> 4)) * 32 + lane) * 16);
        acc[kb & 1] = __builtin_amdgcn_wmma_f32_16x16x32_bf16(
            false, av, false, bv, (short)0, acc[kb & 1], false, false);
    }

    float* yb = Y + (size_t)(R0 + hi * 8) * DOUT + col;
#pragma unroll
    for (int j = 0; j < 8; ++j) yb[(size_t)j * DOUT] = acc[0][j] + acc[1][j];
}

// ---------------------------------------------------------------------------
// LayerNorm over 64 dims, one wave per row (2 dims/lane), optional GELU.
// Safe in-place.
// ---------------------------------------------------------------------------
__global__ void ln_kernel(const float* __restrict__ X, float* __restrict__ Y,
                          const float* __restrict__ g, const float* __restrict__ b,
                          int N, int dogelu) {
    const int wid = blockIdx.x * (blockDim.x >> 5) + (threadIdx.x >> 5);
    if (wid >= N) return;
    const int lane = threadIdx.x & 31, d0 = lane * 2, d1 = d0 + 1;
    const float* x = X + (size_t)wid * HID;
    float x0 = x[d0], x1 = x[d1];
    float s = x0 + x1;
    for (int m = 16; m; m >>= 1) s += __shfl_xor(s, m, 32);
    const float mean = s * (1.0f / HID);
    float e0 = x0 - mean, e1 = x1 - mean;
    float v = e0 * e0 + e1 * e1;
    for (int m = 16; m; m >>= 1) v += __shfl_xor(v, m, 32);
    const float rinv = rsqrtf(v * (1.0f / HID) + 1e-5f);
    float y0 = e0 * rinv * g[d0] + b[d0];
    float y1 = e1 * rinv * g[d1] + b[d1];
    if (dogelu) { y0 = gelu_f(y0); y1 = gelu_f(y1); }
    float* y = Y + (size_t)wid * HID;
    y[d0] = y0; y[d1] = y1;
}

// ---------------------------------------------------------------------------
// Fused encoder + embedding fuse: out = fuse([l2(gelu(ln(l1 x))); emb[id]])
// One wave per node, 2 dims per lane; cross-lane sharing via shuffles.
// ---------------------------------------------------------------------------
__global__ void enc_fuse_kernel(const float* __restrict__ X, int dinx,
                                const float* __restrict__ W1, const float* __restrict__ B1,
                                const float* __restrict__ LG, const float* __restrict__ LBv,
                                const float* __restrict__ W2, const float* __restrict__ B2,
                                const float* __restrict__ Emb, int nmod,
                                const float* __restrict__ WF, const float* __restrict__ BF,
                                float* __restrict__ Out, int N) {
    const int wid = blockIdx.x * (blockDim.x >> 5) + (threadIdx.x >> 5);
    if (wid >= N) return;
    const int lane = threadIdx.x & 31, d0 = lane * 2, d1 = d0 + 1;
    const float* x = X + (size_t)wid * dinx;
    float xv[8];
#pragma unroll
    for (int j = 0; j < 8; ++j) xv[j] = (j < dinx) ? x[j] : 0.0f;

    float t0 = B1[d0], t1 = B1[d1];
#pragma unroll
    for (int j = 0; j < 8; ++j) {
        if (j < dinx) { t0 += xv[j] * W1[j * HID + d0]; t1 += xv[j] * W1[j * HID + d1]; }
    }
    // layernorm
    float s = t0 + t1;
    for (int m = 16; m; m >>= 1) s += __shfl_xor(s, m, 32);
    const float mean = s * (1.0f / HID);
    float e0 = t0 - mean, e1 = t1 - mean;
    float v = e0 * e0 + e1 * e1;
    for (int m = 16; m; m >>= 1) v += __shfl_xor(v, m, 32);
    const float rinv = rsqrtf(v * (1.0f / HID) + 1e-5f);
    const float g0 = gelu_f(e0 * rinv * LG[d0] + LBv[d0]);
    const float g1 = gelu_f(e1 * rinv * LG[d1] + LBv[d1]);
    // l2 (64x64) via shuffle broadcast
    float u0 = B2[d0], u1 = B2[d1];
    for (int k2 = 0; k2 < 32; ++k2) {
        const float glo = __shfl(g0, k2, 32);
        const float ghi = __shfl(g1, k2, 32);
        const int ka = 2 * k2, kb = ka + 1;
        u0 += glo * W2[ka * HID + d0] + ghi * W2[kb * HID + d0];
        u1 += glo * W2[ka * HID + d1] + ghi * W2[kb * HID + d1];
    }
    // fuse (80 -> 64): 64 from u, 16 from embedding
    float f0 = BF[d0], f1 = BF[d1];
    for (int k2 = 0; k2 < 32; ++k2) {
        const float ulo = __shfl(u0, k2, 32);
        const float uhi = __shfl(u1, k2, 32);
        const int ka = 2 * k2, kb = ka + 1;
        f0 += ulo * WF[ka * HID + d0] + uhi * WF[kb * HID + d0];
        f1 += ulo * WF[ka * HID + d1] + uhi * WF[kb * HID + d1];
    }
    const int id = wid % nmod;
    const float* em = Emb + (size_t)id * EMB_D;
#pragma unroll
    for (int j = 0; j < EMB_D; ++j) {
        const float ev = em[j];
        f0 += ev * WF[(HID + j) * HID + d0];
        f1 += ev * WF[(HID + j) * HID + d1];
    }
    Out[(size_t)wid * HID + d0] = f0;
    Out[(size_t)wid * HID + d1] = f1;
}

// ---------------------------------------------------------------------------
// Edge-attr projection: out[E x 64] = ea[E x 4] @ We[4 x 64] + be
// ---------------------------------------------------------------------------
__global__ void eproj_kernel(const float* __restrict__ ea, const float* __restrict__ We,
                             const float* __restrict__ be, float* __restrict__ out, int E) {
    const int gid = blockIdx.x * blockDim.x + threadIdx.x;
    if (gid >= E * HEADS) return;
    const int e = gid >> 4, h = gid & 15;
    const float* a = ea + (size_t)e * 4;
    const float a0 = a[0], a1 = a[1], a2 = a[2], a3 = a[3];
#pragma unroll
    for (int c = 0; c < HC; ++c) {
        const int d = h * HC + c;
        out[(size_t)e * HID + d] = be[d] + a0 * We[0 * HID + d] + a1 * We[1 * HID + d]
                                         + a2 * We[2 * HID + d] + a3 * We[3 * HID + d];
    }
}

// ---------------------------------------------------------------------------
// Attention: thread per (edge, head). alpha = q[dst].kj / sqrt(HC)
// ---------------------------------------------------------------------------
__device__ __forceinline__ float att_alpha(const float* __restrict__ Q,
                                           const float* __restrict__ Km,
                                           const float* __restrict__ EP,
                                           int s, int d, int e, int h) {
    const float* q = Q  + (size_t)d * HID + h * HC;
    const float* k = Km + (size_t)s * HID + h * HC;
    float a = 0.0f;
#pragma unroll
    for (int c = 0; c < HC; ++c) {
        float kv = k[c];
        if (EP) kv += EP[(size_t)e * HID + h * HC + c];
        a += q[c] * kv;
    }
    return a * 0.5f;  // 1/sqrt(4)
}

__global__ void att_amax_kernel(const int* __restrict__ src, const int* __restrict__ dst,
                                const float* __restrict__ Q, const float* __restrict__ Km,
                                const float* __restrict__ EP, unsigned* __restrict__ AM, int E) {
    const int gid = blockIdx.x * blockDim.x + threadIdx.x;
    if (gid >= E * HEADS) return;
    const int e = gid >> 4, h = gid & 15;
    __builtin_prefetch(src + e + 2048, 0, 1);
    const int s = src[e], d = dst[e];
    const float a = att_alpha(Q, Km, EP, s, d, e, h);
    atomicMax(AM + (size_t)d * HEADS + h, fkey(a));
}

__global__ void att_den_kernel(const int* __restrict__ src, const int* __restrict__ dst,
                               const float* __restrict__ Q, const float* __restrict__ Km,
                               const float* __restrict__ EP, const unsigned* __restrict__ AM,
                               float* __restrict__ DN, int E) {
    const int gid = blockIdx.x * blockDim.x + threadIdx.x;
    if (gid >= E * HEADS) return;
    const int e = gid >> 4, h = gid & 15;
    const int s = src[e], d = dst[e];
    const float a  = att_alpha(Q, Km, EP, s, d, e, h);
    const float am = funkey(AM[(size_t)d * HEADS + h]);
    atomicAdd(DN + (size_t)d * HEADS + h, expf(a - am));
}

__global__ void att_out_kernel(const int* __restrict__ src, const int* __restrict__ dst,
                               const float* __restrict__ Q, const float* __restrict__ Km,
                               const float* __restrict__ V, const float* __restrict__ EP,
                               const unsigned* __restrict__ AM, const float* __restrict__ DN,
                               float* __restrict__ O, int E) {
    const int gid = blockIdx.x * blockDim.x + threadIdx.x;
    if (gid >= E * HEADS) return;
    const int e = gid >> 4, h = gid & 15;
    const int s = src[e], d = dst[e];
    const float a  = att_alpha(Q, Km, EP, s, d, e, h);
    const float am = funkey(AM[(size_t)d * HEADS + h]);
    const float wgt = expf(a - am) / (DN[(size_t)d * HEADS + h] + 1e-16f);
#pragma unroll
    for (int c = 0; c < HC; ++c) {
        float vv = V[(size_t)s * HID + h * HC + c];
        if (EP) vv += EP[(size_t)e * HID + h * HC + c];
        atomicAdd(O + (size_t)d * HID + h * HC + c, vv * wgt);
    }
}

// ---------------------------------------------------------------------------
// Misc small kernels
// ---------------------------------------------------------------------------
__global__ void fill32_kernel(unsigned* __restrict__ p, unsigned v, int n) {
    const int i = blockIdx.x * blockDim.x + threadIdx.x;
    if (i < n) p[i] = v;
}

__global__ void add_half_kernel(float* __restrict__ H, const float* __restrict__ A, int n) {
    const int i = blockIdx.x * blockDim.x + threadIdx.x;
    if (i < n) H[i] += 0.5f * A[i];
}

// batched mean over N1 rows per batch: g[b][d] = mean_r h[(b*N1+r)*64+d]
__global__ void pool_reduce_kernel(const float* __restrict__ H, float* __restrict__ G) {
    const int b = blockIdx.x;
    const int tid = threadIdx.x;
    const int d = tid & 63, part = tid >> 6;  // 4 partials per dim
    __shared__ float red[256];
    float s = 0.0f;
    for (int r = part; r < N1C; r += 4)
        s += H[((size_t)(b * N1C + r)) * HID + d];
    red[part * 64 + d] = s;
    __syncthreads();
    if (part == 0)
        G[b * HID + d] = (red[d] + red[64 + d] + red[128 + d] + red[192 + d]) * (1.0f / N1C);
}

// out = gelu(pool([h_row; g_batch]))  (128 -> 64), wave per node
__global__ void pool_mlp_kernel(const float* __restrict__ H, const float* __restrict__ G,
                                const float* __restrict__ WP, const float* __restrict__ BP,
                                float* __restrict__ Out, int N, int nper) {
    const int wid = blockIdx.x * (blockDim.x >> 5) + (threadIdx.x >> 5);
    if (wid >= N) return;
    const int lane = threadIdx.x & 31, d0 = lane * 2, d1 = d0 + 1;
    const int bb = wid / nper;
    const float* h = H + (size_t)wid * HID;
    const float* g = G + (size_t)bb * HID;
    float a0 = BP[d0], a1 = BP[d1];
    for (int k = 0; k < HID; ++k) {
        const float hv = h[k];
        a0 += hv * WP[k * HID + d0]; a1 += hv * WP[k * HID + d1];
    }
    for (int k = 0; k < HID; ++k) {
        const float gv = g[k];
        a0 += gv * WP[(HID + k) * HID + d0]; a1 += gv * WP[(HID + k) * HID + d1];
    }
    Out[(size_t)wid * HID + d0] = gelu_f(a0);
    Out[(size_t)wid * HID + d1] = gelu_f(a1);
}

// out[N x 4] = gelu(T[N x 32]) @ W3[32 x 4] + b3
__global__ void dec_out_kernel(const float* __restrict__ T, const float* __restrict__ W3,
                               const float* __restrict__ B3, float* __restrict__ Out, int N) {
    const int n = blockIdx.x * blockDim.x + threadIdx.x;
    if (n >= N) return;
    const float* t = T + (size_t)n * 32;
    float a0 = B3[0], a1 = B3[1], a2 = B3[2], a3 = B3[3];
#pragma unroll
    for (int k = 0; k < 32; ++k) {
        const float gk = gelu_f(t[k]);
        a0 += gk * W3[k * 4 + 0]; a1 += gk * W3[k * 4 + 1];
        a2 += gk * W3[k * 4 + 2]; a3 += gk * W3[k * 4 + 3];
    }
    float* o = Out + (size_t)n * 4;
    o[0] = a0; o[1] = a1; o[2] = a2; o[3] = a3;
}

// ---------------------------------------------------------------------------
// Host orchestration
// ---------------------------------------------------------------------------
extern "C" void kernel_launch(void* const* d_in, const int* in_sizes, int n_in,
                              void* d_out, int out_size, void* d_ws, size_t ws_size,
                              hipStream_t stream) {
    (void)in_sizes; (void)n_in; (void)out_size; (void)ws_size;
    auto P  = [&](int i) { return (const float*)d_in[i]; };

    const float* x1d = P(0);
    const float* x2d = P(1);
    const int* eiP   = (const int*)d_in[2];   // [2, 16000]
    const int* eiS   = (const int*)d_in[3];   // [2, 800000]
    const int* ei12  = (const int*)d_in[4];   // [2, 8000]
    const int* ei21  = (const int*)d_in[5];   // [2, 8000]
    const float* eaP = P(6);
    const float* ea12 = P(7);
    const float* ea21 = P(8);
    // params flattened (jax tree order: dict keys alphabetical):
    // 9..16 dec1d{l1.b,l1.w,l2.b,l2.w,l3.b,l3.w,ln.b,ln.g}; 17..24 dec2d;
    // 25 emb1d; 26 emb2d; 27..32 enc1d{l1.b,l1.w,l2.b,l2.w,ln.b,ln.g};
    // 33..38 enc2d; 39,40 fn1d{b,g}; 41,42 fn2d; 43,44 fuse1d{b,w};
    // 45,46 fuse2d; 47+L*42 layers[L]{c12(e,k,q,skip,v b/w)=+0..9,
    // c21=+10..19, ln1d{b,g}=+20,21, ln2d=+22,23, pipe=+24..33,
    // surface(k,q,skip,v)=+34..41}; 215 pool.b; 216 pool.w
    float* outp = (float*)d_out;

    // -------- workspace layout --------
    char* wsb = (char*)d_ws;
    size_t off = 0;
    auto alloc = [&](size_t nfl) { float* p = (float*)(wsb + off); off += nfl * sizeof(float); return p; };
    float* h1   = alloc((size_t)N1T * HID);
    float* h2   = alloc((size_t)N2T * HID);
    float* hn1  = alloc((size_t)N1T * HID);
    float* hn2  = alloc((size_t)N2T * HID);
    float* q1   = alloc((size_t)N1T * HID);
    float* k1   = alloc((size_t)N1T * HID);
    float* v1   = alloc((size_t)N1T * HID);
    float* o1   = alloc((size_t)N1T * HID);
    float* q2   = alloc((size_t)N2T * HID);
    float* k2   = alloc((size_t)N2T * HID);
    float* v2   = alloc((size_t)N2T * HID);
    float* o2   = alloc((size_t)N2T * HID);
    unsigned* am1 = (unsigned*)alloc((size_t)N1T * HEADS);
    float*    dn1 = alloc((size_t)N1T * HEADS);
    unsigned* am2 = (unsigned*)alloc((size_t)N2T * HEADS);
    float*    dn2 = alloc((size_t)N2T * HEADS);
    float* ep   = alloc((size_t)E_PIPE_C * HID);   // shared edge-proj scratch
    float* gp   = alloc((size_t)BATCH * HID);

    const dim3 blk(256);
    auto wgrid = [](int n) { return dim3((unsigned)((n + 7) / 8)); };      // wave/node
    auto tgrid = [](long long n) { return dim3((unsigned)((n + 255) / 256)); };

    auto lin64 = [&](const float* X, int iw, int ib, float* Y, int N) {
        linear_wmma_kernel<64, 64><<<dim3((unsigned)(N / 32)), blk, 0, stream>>>(
            X, P(iw), P(ib), Y);
    };
    auto attn = [&](const int* ei, int E, const float* Q, const float* K,
                    const float* V, const float* EP, unsigned* AM, float* DN,
                    float* O, int nd) {
        fill32_kernel<<<tgrid((long long)nd * HEADS), blk, 0, stream>>>(AM, 0u, nd * HEADS);
        fill32_kernel<<<tgrid((long long)nd * HEADS), blk, 0, stream>>>((unsigned*)DN, 0u, nd * HEADS);
        const int* s = ei; const int* d = ei + E;
        att_amax_kernel<<<tgrid((long long)E * HEADS), blk, 0, stream>>>(s, d, Q, K, EP, AM, E);
        att_den_kernel <<<tgrid((long long)E * HEADS), blk, 0, stream>>>(s, d, Q, K, EP, AM, DN, E);
        att_out_kernel <<<tgrid((long long)E * HEADS), blk, 0, stream>>>(s, d, Q, K, V, EP, AM, DN, O, E);
    };

    // -------- encoders + fuse --------
    enc_fuse_kernel<<<wgrid(N1T), blk, 0, stream>>>(x1d, 8, P(28), P(27), P(32), P(31),
                                                    P(30), P(29), P(25), N1C, P(44), P(43), h1, N1T);
    enc_fuse_kernel<<<wgrid(N2T), blk, 0, stream>>>(x2d, 6, P(34), P(33), P(38), P(37),
                                                    P(36), P(35), P(26), N2C, P(46), P(45), h2, N2T);

    // -------- 4 message-passing layers --------
    for (int L = 0; L < 4; ++L) {
        const int LB = 47 + L * 42;
        ln_kernel<<<wgrid(N1T), blk, 0, stream>>>(h1, hn1, P(LB + 21), P(LB + 20), N1T, 0);
        ln_kernel<<<wgrid(N2T), blk, 0, stream>>>(h2, hn2, P(LB + 23), P(LB + 22), N2T, 0);

        // surface (n2 -> n2), no edge attr: k=+34/35 q=+36/37 skip=+38/39 v=+40/41
        lin64(hn2, LB + 37, LB + 36, q2, N2T);
        lin64(hn2, LB + 35, LB + 34, k2, N2T);
        lin64(hn2, LB + 41, LB + 40, v2, N2T);
        lin64(hn2, LB + 39, LB + 38, o2, N2T);   // skip -> accumulator
        attn(eiS, E_SURF_C, q2, k2, v2, nullptr, am2, dn2, o2, N2T);
        add_half_kernel<<<tgrid((long long)N2T * HID), blk, 0, stream>>>(h2, o2, N2T * HID);

        // c12 (n1 -> n2), edge attr: e=+0/1 k=+2/3 q=+4/5 skip=+6/7 v=+8/9
        eproj_kernel<<<tgrid((long long)E_CC * HEADS), blk, 0, stream>>>(ea12, P(LB + 1), P(LB + 0), ep, E_CC);
        lin64(hn2, LB + 5, LB + 4, q2, N2T);
        lin64(hn1, LB + 3, LB + 2, k1, N1T);
        lin64(hn1, LB + 9, LB + 8, v1, N1T);
        lin64(hn2, LB + 7, LB + 6, o2, N2T);
        attn(ei12, E_CC, q2, k1, v1, ep, am2, dn2, o2, N2T);
        add_half_kernel<<<tgrid((long long)N2T * HID), blk, 0, stream>>>(h2, o2, N2T * HID);

        // pipe (n1 -> n1), edge attr: base +24
        eproj_kernel<<<tgrid((long long)E_PIPE_C * HEADS), blk, 0, stream>>>(eaP, P(LB + 25), P(LB + 24), ep, E_PIPE_C);
        lin64(hn1, LB + 29, LB + 28, q1, N1T);
        lin64(hn1, LB + 27, LB + 26, k1, N1T);
        lin64(hn1, LB + 33, LB + 32, v1, N1T);
        lin64(hn1, LB + 31, LB + 30, o1, N1T);
        attn(eiP, E_PIPE_C, q1, k1, v1, ep, am1, dn1, o1, N1T);
        add_half_kernel<<<tgrid((long long)N1T * HID), blk, 0, stream>>>(h1, o1, N1T * HID);

        // c21 (n2 -> n1), edge attr: base +10
        eproj_kernel<<<tgrid((long long)E_CC * HEADS), blk, 0, stream>>>(ea21, P(LB + 11), P(LB + 10), ep, E_CC);
        lin64(hn1, LB + 15, LB + 14, q1, N1T);
        lin64(hn2, LB + 13, LB + 12, k2, N2T);
        lin64(hn2, LB + 19, LB + 18, v2, N2T);
        lin64(hn1, LB + 17, LB + 16, o1, N1T);
        attn(ei21, E_CC, q1, k2, v2, ep, am1, dn1, o1, N1T);
        add_half_kernel<<<tgrid((long long)N1T * HID), blk, 0, stream>>>(h1, o1, N1T * HID);
    }

    // -------- final norms, pooling --------
    ln_kernel<<<wgrid(N1T), blk, 0, stream>>>(h1, h1, P(40), P(39), N1T, 0);
    ln_kernel<<<wgrid(N2T), blk, 0, stream>>>(h2, h2, P(42), P(41), N2T, 0);
    pool_reduce_kernel<<<dim3(BATCH), blk, 0, stream>>>(h1, gp);
    pool_mlp_kernel<<<wgrid(N1T), blk, 0, stream>>>(h1, gp, P(216), P(215), hn1, N1T, N1C);

    // -------- decoder 1d: out[0 .. 8000*4) --------
    lin64(hn1, 10, 9, q1, N1T);
    ln_kernel<<<wgrid(N1T), blk, 0, stream>>>(q1, k1, P(16), P(15), N1T, 1);
    linear_wmma_kernel<64, 32><<<dim3((unsigned)(N1T / 64)), blk, 0, stream>>>(
        k1, P(12), P(11), v1);
    dec_out_kernel<<<tgrid(N1T), blk, 0, stream>>>(v1, P(14), P(13), outp, N1T);

    // -------- decoder 2d: out[8000*4 ..) --------
    lin64(h2, 18, 17, q2, N2T);
    ln_kernel<<<wgrid(N2T), blk, 0, stream>>>(q2, k2, P(24), P(23), N2T, 1);
    linear_wmma_kernel<64, 32><<<dim3((unsigned)(N2T / 64)), blk, 0, stream>>>(
        k2, P(20), P(19), v2);
    dec_out_kernel<<<tgrid(N2T), blk, 0, stream>>>(v2, P(22), P(21), outp + (size_t)N1T * KOUT, N2T);
}
